// MOPI_HFRS_Light_2748779070014
// MI455X (gfx1250) — matrix operand order repaired and checked by reading, not compiled, not run on hardware
//
#include <hip/hip_runtime.h>

// ---------------- problem constants (match reference) ----------------
#define U_NUM     50000
#define F_NUM     20000
#define N_TOT     70000      // U_NUM + F_NUM
#define D_EMB     128
#define FEAT_DIM  256
#define E_EDGES   600000     // unique user-food edges (directed = 2*E_EDGES)
#define N_HEADS   4
#define D_HEAD    32
#define SIM_THRESH 0.3f

// ---------------- CDNA5 WMMA types ----------------
typedef __bf16 bf16_t;
typedef bf16_t v16bf __attribute__((ext_vector_type(16)));
typedef float  v8f   __attribute__((ext_vector_type(8)));

union ABFrag { v16bf v; uint4 q[2]; };

__device__ __forceinline__ unsigned short f32_to_bf16_rne(float f) {
  unsigned int x = __float_as_uint(f);
  x += 0x7fffu + ((x >> 16) & 1u);   // round-to-nearest-even
  return (unsigned short)(x >> 16);
}

// ---------------- small utility kernels ----------------
__global__ void k_cvt_bf16(const float* __restrict__ in,
                           unsigned short* __restrict__ out, int n) {
  int i = blockIdx.x * blockDim.x + threadIdx.x;
  if (i < n) out[i] = f32_to_bf16_rne(in[i]);
}

// W [K][N] f32 row-major -> Wt [N][K] bf16 row-major
__global__ void k_transpose_bf16(const float* __restrict__ in,
                                 unsigned short* __restrict__ out, int K, int N) {
  int i = blockIdx.x * blockDim.x + threadIdx.x;
  if (i >= K * N) return;
  int k = i / N, n = i - k * N;
  out[n * K + k] = f32_to_bf16_rne(in[i]);
}

__global__ void k_zero(float* __restrict__ p, int n) {
  int i = blockIdx.x * blockDim.x + threadIdx.x;
  if (i < n) p[i] = 0.f;
}

__global__ void k_copy(const float* __restrict__ src, float* __restrict__ dst, int n) {
  int i = blockIdx.x * blockDim.x + threadIdx.x;
  if (i < n) dst[i] = src[i];
}

__global__ void k_acc_add(float* __restrict__ acc, const float* __restrict__ x, int n) {
  int i = blockIdx.x * blockDim.x + threadIdx.x;
  if (i < n) acc[i] += x[i];
}

__global__ void k_scale(float* __restrict__ p, float s, int n) {
  int i = blockIdx.x * blockDim.x + threadIdx.x;
  if (i < n) p[i] *= s;
}

// ---------------- WMMA GEMM: C[M,128] = act(A[M,K] @ B[K,128] + bias) ----------------
// A: bf16 row-major [M,K].  Bt: bf16 [128,K] (pre-transposed).  One wave per 16-row strip,
// full N=128 per wave: A fragment loaded once per K-step, 8 v_wmma per K-step.
__global__ __launch_bounds__(32) void k_gemm_n128(
    const unsigned short* __restrict__ A,
    const unsigned short* __restrict__ Bt,
    const float* __restrict__ bias,       // [128] or nullptr
    float* __restrict__ C,                // [M,128] f32 or nullptr
    unsigned short* __restrict__ Cb,      // [M,128] bf16 or nullptr
    int K, int relu)
{
  const int lane  = threadIdx.x;
  const int nl    = lane & 15;
  const int hi    = lane >> 4;           // 0: lanes 0-15, 1: lanes 16-31
  const int m0    = blockIdx.x * 16;
  const int arow  = m0 + nl;
  const int kbase = hi * 16;             // A: K chunk  [kbase, kbase+16) contiguous
  const int kbB   = hi * 16;             // B: K chunk  [kbB, kbB+16) contiguous

  v8f acc[8];
#pragma unroll
  for (int t = 0; t < 8; ++t)
    acc[t] = (v8f){0.f, 0.f, 0.f, 0.f, 0.f, 0.f, 0.f, 0.f};

  for (int k0 = 0; k0 < K; k0 += 32) {
    ABFrag a;
    const unsigned short* ap = A + (size_t)arow * K + (k0 + kbase);
    a.q[0] = *(const uint4*)(ap);
    a.q[1] = *(const uint4*)(ap + 8);
#pragma unroll
    for (int t = 0; t < 8; ++t) {
      ABFrag b;
      const unsigned short* bp = Bt + (size_t)(t * 16 + nl) * K + (k0 + kbB);
      b.q[0] = *(const uint4*)(bp);
      b.q[1] = *(const uint4*)(bp + 8);
      acc[t] = __builtin_amdgcn_wmma_f32_16x16x32_bf16(
          /*neg_a=*/false, a.v, /*neg_b=*/false, b.v,
          /*c_mod=*/(short)0, acc[t], /*reuse_a=*/false, /*reuse_b=*/false);
    }
  }

  const int mrow = m0 + hi * 8;          // C: lane writes rows mrow..mrow+7, column nl of tile
#pragma unroll
  for (int t = 0; t < 8; ++t) {
    const int col = t * 16 + nl;
    const float bv = bias ? bias[col] : 0.f;
#pragma unroll
    for (int r = 0; r < 8; ++r) {
      float v = acc[t][r] + bv;
      if (relu) v = fmaxf(v, 0.f);
      size_t idx = (size_t)(mrow + r) * D_EMB + col;
      if (C)  C[idx]  = v;
      if (Cb) Cb[idx] = f32_to_bf16_rne(v);
    }
  }
}

// ---------------- per-edge mask + degree ----------------
__global__ void k_edge_mask(const int* __restrict__ ei,        // [2, 2E] flat
                            const float* __restrict__ Q,       // [U,128]
                            const float* __restrict__ Km,      // [F,128]
                            const float* __restrict__ cw,      // [2]
                            float* __restrict__ mask,          // [E]
                            float* __restrict__ deg,           // [N_TOT] (pre-zeroed)
                            int2* __restrict__ ends)           // [E] (u, f)
{
  int e = blockIdx.x * blockDim.x + threadIdx.x;
  if (e >= E_EDGES) return;
  int u = ei[e];                         // first half: row = user index
  int f = ei[2 * E_EDGES + e] - U_NUM;   // first half: col = food index + U

  float c0 = cw[0], c1 = cw[1];
  float mx = fmaxf(c0, c1);
  float e0 = __expf(c0 - mx), e1 = __expf(c1 - mx);
  float w0 = e0 / (e0 + e1), w1 = 1.f - w0;

  const float* q = Q  + (size_t)u * D_EMB;
  const float* k = Km + (size_t)f * D_EMB;
  float simsum = 0.f;
#pragma unroll
  for (int h = 0; h < N_HEADS; ++h) {
    float dot = 0.f, nq = 0.f, nk = 0.f;
#pragma unroll
    for (int i = 0; i < D_HEAD; i += 4) {
      float4 qa = *(const float4*)(q + h * D_HEAD + i);
      float4 ka = *(const float4*)(k + h * D_HEAD + i);
      dot += qa.x * ka.x + qa.y * ka.y + qa.z * ka.z + qa.w * ka.w;
      nq  += qa.x * qa.x + qa.y * qa.y + qa.z * qa.z + qa.w * qa.w;
      nk  += ka.x * ka.x + ka.y * ka.y + ka.z * ka.z + ka.w * ka.w;
    }
    simsum += dot / (sqrtf(nq) * sqrtf(nk) + 1e-8f);
  }
  float mf    = (simsum * 0.25f > SIM_THRESH) ? 1.f : 0.f;
  float fused = w0 + w1 * mf;
  float m     = (fused > 0.5f) ? 1.f : 0.f;
  mask[e] = m;
  ends[e] = make_int2(u, f);
  if (m != 0.f) {
    atomicAdd(&deg[u], 1.f);
    atomicAdd(&deg[U_NUM + f], 1.f);
  }
}

__global__ void k_norm(const int2* __restrict__ ends, const float* __restrict__ mask,
                       const float* __restrict__ deg, float* __restrict__ norm) {
  int e = blockIdx.x * blockDim.x + threadIdx.x;
  if (e >= E_EDGES) return;
  int2 uf = ends[e];
  float du = fmaxf(deg[uf.x], 1.f);
  float df = fmaxf(deg[U_NUM + uf.y], 1.f);
  norm[e] = mask[e] * rsqrtf(du * df);
}

// ---------------- LightGCN scatter: both directions of each symmetric edge ----------------
// thread t: edge e = t>>5, float4 chunk c = (t&31)*4.  L2-resident x buffers (35.8 MB << 192 MB).
__global__ void k_scatter(const int2* __restrict__ ends, const float* __restrict__ norm,
                          const float* __restrict__ xc, float* __restrict__ xn) {
  int t = blockIdx.x * blockDim.x + threadIdx.x;
  int e = t >> 5;
  if (e >= E_EDGES) return;
  float ne = norm[e];
  if (ne == 0.f) return;
  int c = (t & 31) * 4;
  int2 uf = ends[e];
  size_t ub = (size_t)uf.x * D_EMB + c;
  size_t fb = (size_t)(U_NUM + uf.y) * D_EMB + c;
  float4 xf = *(const float4*)(xc + fb);
  float4 xu = *(const float4*)(xc + ub);
  atomicAdd(&xn[ub + 0], ne * xf.x);
  atomicAdd(&xn[ub + 1], ne * xf.y);
  atomicAdd(&xn[ub + 2], ne * xf.z);
  atomicAdd(&xn[ub + 3], ne * xf.w);
  atomicAdd(&xn[fb + 0], ne * xu.x);
  atomicAdd(&xn[fb + 1], ne * xu.y);
  atomicAdd(&xn[fb + 2], ne * xu.z);
  atomicAdd(&xn[fb + 3], ne * xu.w);
}

static inline int cdiv(int a, int b) { return (a + b - 1) / b; }

extern "C" void kernel_launch(void* const* d_in, const int* in_sizes, int n_in,
                              void* d_out, int out_size, void* d_ws, size_t ws_size,
                              hipStream_t stream) {
  const float* user_feat  = (const float*)d_in[0];
  const float* food_feat  = (const float*)d_in[1];
  const int*   edge_index = (const int*)d_in[2];
  const float* W_user     = (const float*)d_in[3];
  const float* b_user     = (const float*)d_in[4];
  const float* W_food     = (const float*)d_in[5];
  const float* b_food     = (const float*)d_in[6];
  const float* Wq         = (const float*)d_in[7];
  const float* Wk         = (const float*)d_in[8];
  const float* cw         = (const float*)d_in[9];
  const float* user_embed = (const float*)d_in[10];
  const float* item_embed = (const float*)d_in[11];
  float* out = (float*)d_out;
  char*  ws  = (char*)d_ws;

  // -------- workspace layout (all offsets 256B-aligned, phases alias) --------
  // Region A [0, 36MB): phase1 = user/food feat bf16; phase2 = x ping buffer
  // Region B [36MB, 72MB): phase2 = x pong buffer
  const size_t offRegA = 0;
  const size_t offRegB = 36000000;
  const size_t offQ    = 72000000;            // Q f32  [U,128]  25.6 MB
  const size_t offKm   = 97600000;            // K f32  [F,128]  10.24 MB
  const size_t offUeB  = 107840000;           // user_emb bf16   12.8 MB
  const size_t offFeB  = 120640000;           // food_emb bf16   5.12 MB
  const size_t offMask = 125760000;           // mask f32 [E]    2.4 MB
  const size_t offNorm = 128160000;           // norm f32 [E]    2.4 MB
  const size_t offDeg  = 130560000;           // deg f32 [N]     0.28 MB
  const size_t offEnds = 130840064;           // int2 [E]        4.8 MB
  const size_t offWuT  = 135640064;           // Wt bf16 buffers
  const size_t offWfT  = offWuT + 65536;
  const size_t offWqT  = offWfT + 65536;
  const size_t offWkT  = offWqT + 32768;

  unsigned short* uf_b = (unsigned short*)(ws + offRegA);
  unsigned short* ff_b = (unsigned short*)(ws + offRegA + (size_t)U_NUM * FEAT_DIM * 2);
  unsigned short* WuT  = (unsigned short*)(ws + offWuT);
  unsigned short* WfT  = (unsigned short*)(ws + offWfT);
  unsigned short* WqT  = (unsigned short*)(ws + offWqT);
  unsigned short* WkT  = (unsigned short*)(ws + offWkT);
  unsigned short* ue_b = (unsigned short*)(ws + offUeB);
  unsigned short* fe_b = (unsigned short*)(ws + offFeB);
  float* Qm   = (float*)(ws + offQ);
  float* Km   = (float*)(ws + offKm);
  float* mask = (float*)(ws + offMask);
  float* norm = (float*)(ws + offNorm);
  float* deg  = (float*)(ws + offDeg);
  int2*  ends = (int2*)(ws + offEnds);
  float* x_a  = (float*)(ws + offRegA);   // aliases feat bf16 (dead after GEMMs)
  float* x_b  = (float*)(ws + offRegB);

  const int T = 256;

  // ---- phase 1: stage features/weights in bf16 ----
  k_cvt_bf16<<<cdiv(U_NUM * FEAT_DIM, T), T, 0, stream>>>(user_feat, uf_b, U_NUM * FEAT_DIM);
  k_cvt_bf16<<<cdiv(F_NUM * FEAT_DIM, T), T, 0, stream>>>(food_feat, ff_b, F_NUM * FEAT_DIM);
  k_transpose_bf16<<<cdiv(FEAT_DIM * D_EMB, T), T, 0, stream>>>(W_user, WuT, FEAT_DIM, D_EMB);
  k_transpose_bf16<<<cdiv(FEAT_DIM * D_EMB, T), T, 0, stream>>>(W_food, WfT, FEAT_DIM, D_EMB);
  k_transpose_bf16<<<cdiv(D_EMB * D_EMB, T), T, 0, stream>>>(Wq, WqT, D_EMB, D_EMB);
  k_transpose_bf16<<<cdiv(D_EMB * D_EMB, T), T, 0, stream>>>(Wk, WkT, D_EMB, D_EMB);

  // ---- phase 2: WMMA GEMMs ----
  // user_emb = relu(user_feat @ W_user + b_user) -> bf16 only (f32 never consumed)
  k_gemm_n128<<<U_NUM / 16, 32, 0, stream>>>(uf_b, WuT, b_user, nullptr, ue_b, FEAT_DIM, 1);
  k_gemm_n128<<<F_NUM / 16, 32, 0, stream>>>(ff_b, WfT, b_food, nullptr, fe_b, FEAT_DIM, 1);
  // Q = user_emb @ Wq, K = food_emb @ Wk (f32 out for cosine sim)
  k_gemm_n128<<<U_NUM / 16, 32, 0, stream>>>(ue_b, WqT, nullptr, Qm, nullptr, D_EMB, 0);
  k_gemm_n128<<<F_NUM / 16, 32, 0, stream>>>(fe_b, WkT, nullptr, Km, nullptr, D_EMB, 0);

  // ---- phase 3: edge mask, degrees, norm ----
  k_zero<<<cdiv(N_TOT, T), T, 0, stream>>>(deg, N_TOT);
  k_edge_mask<<<cdiv(E_EDGES, T), T, 0, stream>>>(edge_index, Qm, Km, cw, mask, deg, ends);
  k_norm<<<cdiv(E_EDGES, T), T, 0, stream>>>(ends, mask, deg, norm);

  // ---- phase 4: LightGCN propagation; acc lives directly in d_out[0 : N*D] ----
  const int ND = N_TOT * D_EMB;
  k_copy<<<cdiv(U_NUM * D_EMB, T), T, 0, stream>>>(user_embed, x_a, U_NUM * D_EMB);
  k_copy<<<cdiv(F_NUM * D_EMB, T), T, 0, stream>>>(item_embed, x_a + (size_t)U_NUM * D_EMB, F_NUM * D_EMB);
  k_copy<<<cdiv(U_NUM * D_EMB, T), T, 0, stream>>>(user_embed, out, U_NUM * D_EMB);
  k_copy<<<cdiv(F_NUM * D_EMB, T), T, 0, stream>>>(item_embed, out + (size_t)U_NUM * D_EMB, F_NUM * D_EMB);

  float* xc = x_a;
  float* xn = x_b;
  for (int layer = 0; layer < 3; ++layer) {
    k_zero<<<cdiv(ND, T), T, 0, stream>>>(xn, ND);
    k_scatter<<<cdiv(E_EDGES * 32, T), T, 0, stream>>>(ends, norm, xc, xn);
    k_acc_add<<<cdiv(ND, T), T, 0, stream>>>(out, xn, ND);
    float* tmp = xc; xc = xn; xn = tmp;
  }
  k_scale<<<cdiv(ND, T), T, 0, stream>>>(out, 0.25f, ND);

  // ---- phase 5: pass-through outputs (user_embed, item_embed) ----
  float* tail = out + (size_t)ND;
  k_copy<<<cdiv(U_NUM * D_EMB, T), T, 0, stream>>>(user_embed, tail, U_NUM * D_EMB);
  k_copy<<<cdiv(F_NUM * D_EMB, T), T, 0, stream>>>(item_embed, tail + (size_t)U_NUM * D_EMB, F_NUM * D_EMB);
}
